// ExpanderLinearLayer_23965917512071
// MI455X (gfx1250) — compile-verified, matrix operand order checked by reading, not run on hardware
//
#include <hip/hip_runtime.h>
#include <hip/hip_bf16.h>

typedef __attribute__((ext_vector_type(16))) __bf16 v16bf;
typedef __attribute__((ext_vector_type(8)))  float  v8f;
typedef __attribute__((ext_vector_type(4)))  int    v4i;

#define BATCH  4096
#define INDIM  1024
#define OUTDIM 1024

#define BM 128
#define BN 128
#define BK 32
#define LDW 40   // padded LDS row stride in bf16 elements: 80 B row = 5x16 B (b128-aligned chunks)

// ---------- bf16 helpers (RNE, bit-exact) ----------
__device__ __forceinline__ unsigned short f32_to_bf16_bits(float f) {
  union { float f; unsigned u; } v; v.f = f;
  unsigned r = v.u + 0x7FFFu + ((v.u >> 16) & 1u);
  return (unsigned short)(r >> 16);
}
__device__ __forceinline__ float bf16_bits_to_f32(unsigned short hs) {
  union { unsigned u; float f; } v; v.u = ((unsigned)hs) << 16;
  return v.f;
}

// ---------- async global->LDS copy (16 B per lane), hedged across toolchains ----------
__device__ __forceinline__ void async_copy16(const void* gp, void* lp) {
#if defined(__has_builtin) && __has_builtin(__builtin_amdgcn_global_load_async_to_lds_b128)
  typedef __attribute__((address_space(1))) v4i* gas_p;  // global int4*
  typedef __attribute__((address_space(3))) v4i* las_p;  // LDS int4*
  __builtin_amdgcn_global_load_async_to_lds_b128((gas_p)gp, (las_p)lp, 0, 0);
#else
  typedef __attribute__((address_space(3))) void* las_vp;
  unsigned lds_off = (unsigned)(unsigned long long)(las_vp)lp;
  asm volatile("global_load_async_to_lds_b128 %0, %1, off"
               :: "v"(lds_off), "v"(gp) : "memory");
#endif
}
__device__ __forceinline__ void wait_async0() {
#if defined(__has_builtin) && __has_builtin(__builtin_amdgcn_s_wait_asynccnt)
  __builtin_amdgcn_s_wait_asynccnt(0);
#else
  asm volatile("s_wait_asynccnt 0" ::: "memory");
#endif
}

// ---------- kernel 1: zero the dense f32 weight scratch ----------
__global__ void expander_zero_w_kernel(float* __restrict__ w, int n) {
  int i = blockIdx.x * blockDim.x + threadIdx.x;
  if (i < n) w[i] = 0.0f;
}

// ---------- kernel 2: COO -> dense Wt[N=ind_out][K=ind_in] (duplicates accumulate) ----------
__global__ void expander_scatter_w_kernel(const float* __restrict__ wgt,
                                          const int* __restrict__ ind_in,
                                          const int* __restrict__ ind_out,
                                          float* __restrict__ w, int nnz) {
  int k = blockIdx.x * blockDim.x + threadIdx.x;
  if (k < nnz) {
    atomicAdd(&w[ind_out[k] * INDIM + ind_in[k]], wgt[k]);
  }
}

// ---------- kernel 3: f32 -> (bf16 hi, bf16 lo) split, done ONCE per array ----------
__global__ void expander_split_kernel(const float* __restrict__ src,
                                      unsigned short* __restrict__ hi,
                                      unsigned short* __restrict__ lo, int n) {
  int i = blockIdx.x * blockDim.x + threadIdx.x;
  if (i < n) {
    float x = src[i];
    unsigned short h = f32_to_bf16_bits(x);
    hi[i] = h;
    lo[i] = f32_to_bf16_bits(x - bf16_bits_to_f32(h));
  }
}

// ---------- fragment loader: wave32 16-bit A/B layout ----------
// A 16x32 (MxK): row = lane&15, element h holds K = h + 8*[h>=8] + 8*[lane>=16]
// B 32x16 (KxN): identical map with N = lane&15; tile stored [N][K] in LDS.
__device__ __forceinline__ v16bf load_frag(const __bf16* __restrict__ base, int lane) {
  const int r    = lane & 15;
  const int koff = (lane & 16) ? 8 : 0;
  const __bf16* row = base + r * LDW;
  v16bf f;
#pragma unroll
  for (int j = 0; j < 8; ++j) {
    const int k = 2 * j + (j >= 4 ? 8 : 0) + koff;  // even -> pair (k, k+1) contiguous
    f[2 * j]     = row[k];
    f[2 * j + 1] = row[k + 1];
  }
  return f;
}

// ---------- kernel 4: out = X @ Wt^T + bias via bf16x3 WMMA, async-LDS staging ----------
__global__ __launch_bounds__(256) void expander_gemm_kernel(
    const unsigned short* __restrict__ x_hi,  // [BATCH][INDIM] bf16 bits
    const unsigned short* __restrict__ x_lo,
    const unsigned short* __restrict__ w_hi,  // [OUTDIM][INDIM] bf16 bits (N-major)
    const unsigned short* __restrict__ w_lo,
    const float* __restrict__ bias,           // [OUTDIM]
    float* __restrict__ out) {                // [BATCH][OUTDIM]
  __shared__ __attribute__((aligned(16))) __bf16 sx_hi[BM * LDW];
  __shared__ __attribute__((aligned(16))) __bf16 sx_lo[BM * LDW];
  __shared__ __attribute__((aligned(16))) __bf16 sw_hi[BN * LDW];
  __shared__ __attribute__((aligned(16))) __bf16 sw_lo[BN * LDW];

  const int tid    = threadIdx.x;
  const int lane   = tid & 31;
  const int wave   = tid >> 5;
  const int wave_m = wave & 3;   // 4 waves along M, 32 rows each
  const int wave_n = wave >> 2;  // 2 waves along N, 64 cols each

  const int n_base = blockIdx.x * BN;
  const int m_base = blockIdx.y * BM;

  v8f acc[2][4];
#pragma unroll
  for (int mt = 0; mt < 2; ++mt)
#pragma unroll
    for (int nt = 0; nt < 4; ++nt)
      acc[mt][nt] = (v8f){0.f, 0.f, 0.f, 0.f, 0.f, 0.f, 0.f, 0.f};

  for (int k0 = 0; k0 < INDIM; k0 += BK) {
    // stage 4 tiles of 128x32 bf16 = 8 KB each via async b128 copies:
    // 512 chunks of 16 B per tile; 256 threads -> 2 chunks/thread/tile.
#pragma unroll
    for (int p = 0; p < 2; ++p) {
      const int e = tid + p * 256;
      const int r = e >> 2;          // tile row 0..127
      const int q = e & 3;           // 16 B chunk within 32-element row
      const int gx = (m_base + r) * INDIM + k0 + q * 8;   // element index
      const int gw = (n_base + r) * INDIM + k0 + q * 8;
      const int ls = r * LDW + q * 8;                     // element index in LDS
      async_copy16(&x_hi[gx], &sx_hi[ls]);
      async_copy16(&x_lo[gx], &sx_lo[ls]);
      async_copy16(&w_hi[gw], &sw_hi[ls]);
      async_copy16(&w_lo[gw], &sw_lo[ls]);
    }
    wait_async0();
    __syncthreads();

    v16bf a_hi[2], a_lo[2], b_hi[4], b_lo[4];
#pragma unroll
    for (int mt = 0; mt < 2; ++mt) {
      a_hi[mt] = load_frag(&sx_hi[(wave_m * 32 + mt * 16) * LDW], lane);
      a_lo[mt] = load_frag(&sx_lo[(wave_m * 32 + mt * 16) * LDW], lane);
    }
#pragma unroll
    for (int nt = 0; nt < 4; ++nt) {
      b_hi[nt] = load_frag(&sw_hi[(wave_n * 64 + nt * 16) * LDW], lane);
      b_lo[nt] = load_frag(&sw_lo[(wave_n * 64 + nt * 16) * LDW], lane);
    }

#pragma unroll
    for (int mt = 0; mt < 2; ++mt)
#pragma unroll
      for (int nt = 0; nt < 4; ++nt) {
        // bf16x3: hi*hi + hi*lo + lo*hi (lo*lo below noise floor)
        acc[mt][nt] = __builtin_amdgcn_wmma_f32_16x16x32_bf16(
            false, a_hi[mt], false, b_hi[nt], (short)0, acc[mt][nt], false, false);
        acc[mt][nt] = __builtin_amdgcn_wmma_f32_16x16x32_bf16(
            false, a_hi[mt], false, b_lo[nt], (short)0, acc[mt][nt], false, false);
        acc[mt][nt] = __builtin_amdgcn_wmma_f32_16x16x32_bf16(
            false, a_lo[mt], false, b_hi[nt], (short)0, acc[mt][nt], false, false);
      }
    __syncthreads();
  }

  // epilogue: C/D layout — element r: M = r + 8*[lane>=16], N = lane&15
  const int mrow = (lane & 16) ? 8 : 0;
  const int ncol = lane & 15;
#pragma unroll
  for (int nt = 0; nt < 4; ++nt) {
    const int N  = n_base + wave_n * 64 + nt * 16 + ncol;
    const float bv = bias[N];
#pragma unroll
    for (int mt = 0; mt < 2; ++mt) {
      const int Mb = m_base + wave_m * 32 + mt * 16 + mrow;
#pragma unroll
      for (int rr = 0; rr < 8; ++rr) {
        out[(Mb + rr) * OUTDIM + N] = acc[mt][nt][rr] + bv;
      }
    }
  }
}

extern "C" void kernel_launch(void* const* d_in, const int* in_sizes, int n_in,
                              void* d_out, int out_size, void* d_ws, size_t ws_size,
                              hipStream_t stream) {
  const float* input   = (const float*)d_in[0];  // [4096*1024] f32
  const float* weight  = (const float*)d_in[1];  // [NNZ] f32
  const float* bias    = (const float*)d_in[2];  // [1024] f32
  const int*   ind_in  = (const int*)d_in[3];    // [NNZ] i32
  const int*   ind_out = (const int*)d_in[4];    // [NNZ] i32
  float*       out     = (float*)d_out;          // [4096*1024] f32
  const int    nnz     = in_sizes[1];

  // ws layout (24 MB total, all 16 B aligned offsets):
  //  [0,4M)   W_f32 [OUTDIM][INDIM]
  //  [4M,6M)  Wt_hi bf16, [6M,8M) Wt_lo bf16
  //  [8M,16M) X_hi bf16,  [16M,24M) X_lo bf16
  char* ws = (char*)d_ws;
  float*          w_f32 = (float*)ws;
  unsigned short* wt_hi = (unsigned short*)(ws + (4u << 20));
  unsigned short* wt_lo = (unsigned short*)(ws + (6u << 20));
  unsigned short* x_hi  = (unsigned short*)(ws + (8u << 20));
  unsigned short* x_lo  = (unsigned short*)(ws + (16u << 20));

  const int wcount = OUTDIM * INDIM;
  const int xcount = BATCH * INDIM;

  expander_zero_w_kernel<<<(wcount + 255) / 256, 256, 0, stream>>>(w_f32, wcount);
  expander_scatter_w_kernel<<<(nnz + 255) / 256, 256, 0, stream>>>(
      weight, ind_in, ind_out, w_f32, nnz);
  expander_split_kernel<<<(wcount + 255) / 256, 256, 0, stream>>>(w_f32, wt_hi, wt_lo, wcount);
  expander_split_kernel<<<(xcount + 255) / 256, 256, 0, stream>>>(input, x_hi, x_lo, xcount);

  dim3 grid(OUTDIM / BN, BATCH / BM);  // 8 x 32 = 256 workgroups
  expander_gemm_kernel<<<grid, 256, 0, stream>>>(x_hi, x_lo, wt_hi, wt_lo, bias, out);
}
